// Decoder_40441412059470
// MI455X (gfx1250) — matrix operand order repaired and checked by reading, not compiled, not run on hardware
//
#include <hip/hip_runtime.h>
#include <math.h>

// ---------------- constants (match reference) ----------------
#define Vv    1000
#define VPAD  1024
#define Ee    512
#define Hh    1024
#define Bb    256
#define Ss    48
#define Tt    45
#define NEGC  99999.0f
#define EPSC  1e-7f

typedef __attribute__((ext_vector_type(16))) __bf16 v16bf;
typedef __attribute__((ext_vector_type(8)))  float  v8f;

// ---------------- async global -> LDS staging (CDNA5 ASYNCcnt path) ----------------
// Copies 64 contiguous bytes per thread from global to LDS without touching VGPRs.
// INST_OFFSET applies to both the LDS and global addresses (ISA 10.7 pseudocode),
// so one (lds, global) base pair covers all four b128 transfers.
__device__ __forceinline__ void async_copy64(const __bf16* g, __bf16* l) {
    const unsigned long long ga = (unsigned long long)(size_t)g;
    const unsigned int       la = (unsigned int)(size_t)l;   // LDS addr = generic[31:0]
    asm volatile("global_load_async_to_lds_b128 %0, %1, off"           :: "v"(la), "v"(ga) : "memory");
    asm volatile("global_load_async_to_lds_b128 %0, %1, off offset:16" :: "v"(la), "v"(ga) : "memory");
    asm volatile("global_load_async_to_lds_b128 %0, %1, off offset:32" :: "v"(la), "v"(ga) : "memory");
    asm volatile("global_load_async_to_lds_b128 %0, %1, off offset:48" :: "v"(la), "v"(ga) : "memory");
}
__device__ __forceinline__ void wait_async0() {
    asm volatile("s_wait_asynccnt 0x0" ::: "memory");
}

// ---------------- WMMA GEMM with async-LDS double-buffered weights ----------------
// C[M,N] = A0[M,K0] * W0[N,K0]^T  (+ A1[M,K1] * W1[N,K1]^T if A1 != null)  + bias
// A, W bf16 row-major (K contiguous). C fp32 row-major.
// Block tile: 128(M) x 128(N). 256 threads = 8 waves; each wave: 16 rows x 8 n-tiles.
// K stages of 64; W tile (128 x 64 bf16) DMA'd into LDS, double buffered.
#define KC      64
#define LDSROW  72   // 64 bf16 + 8 pad per row (144 B) to spread LDS banks

__global__ __launch_bounds__(256)
void gemm2_bf16_wmma(float* __restrict__ C,
                     const __bf16* __restrict__ A0, const __bf16* __restrict__ W0, int K0,
                     const __bf16* __restrict__ A1, const __bf16* __restrict__ W1, int K1,
                     const float* __restrict__ bias, int N)
{
    __shared__ __bf16 lds[2][128 * LDSROW];

    const int tid  = threadIdx.x;
    const int lane = tid & 31;
    const int wave = tid >> 5;
    const int half = lane >> 4;     // 0: lanes 0-15, 1: lanes 16-31
    const int l16  = lane & 15;
    const int mBase = blockIdx.y * 128 + wave * 16;
    const int nBase = blockIdx.x * 128;

    v8f acc[8];
#pragma unroll
    for (int i = 0; i < 8; i++)
#pragma unroll
        for (int r = 0; r < 8; r++) acc[i][r] = 0.0f;

    // cooperative staging role: 2 threads per n-row, each handles 32 k (64 B)
    const int srow = tid >> 1;
    const int skh  = tid & 1;
    __bf16* lslot0 = &lds[0][srow * LDSROW + skh * 32];
    __bf16* lslot1 = &lds[1][srow * LDSROW + skh * 32];

#pragma unroll 1
    for (int mat = 0; mat < 2; mat++) {
        const __bf16* A = mat ? A1 : A0;
        const __bf16* W = mat ? W1 : W0;
        const int K = mat ? K1 : K0;
        if (A == nullptr || K == 0) break;

        const int S = K / KC;
        // A fragment base: row (mBase+l16); lanes>=16 take K offsets +8/+24
        const __bf16* aBase  = A + (size_t)(mBase + l16) * K + half * 8;
        const __bf16* wStage = W + (size_t)(nBase + srow) * K + skh * 32;

        // prologue: DMA stage 0 into buffer 0
        async_copy64(wStage, lslot0);
        wait_async0();
        __syncthreads();

        for (int s = 0; s < S; s++) {
            const int cur = s & 1;
            if (s + 1 < S)   // fire next stage's DMA; overlaps with the 16 WMMAs below
                async_copy64(wStage + (size_t)(s + 1) * KC, cur ? lslot0 : lslot1);

#pragma unroll
            for (int kc = 0; kc < 2; kc++) {
                union U { v16bf v; uint4 u[2]; };
                U a;
                const uint4* ap = reinterpret_cast<const uint4*>(aBase + s * KC + kc * 32);
                a.u[0] = ap[0];   // K: kf+0 .. kf+7
                a.u[1] = ap[2];   // K: kf+16 .. kf+23
#pragma unroll
                for (int nt = 0; nt < 8; nt++) {
                    U b;
                    const uint4* bp = reinterpret_cast<const uint4*>(
                        &lds[cur][(nt * 16 + l16) * LDSROW + kc * 32 + half * 8]);
                    b.u[0] = bp[0];   // K: kb+0 .. kb+7
                    b.u[1] = bp[2];   // K: kb+16 .. kb+23
                    acc[nt] = __builtin_amdgcn_wmma_f32_16x16x32_bf16(
                        false, a.v, false, b.v, (short)0, acc[nt], false, false);
                }
            }
            if (s + 1 < S) wait_async0();
            __syncthreads();
        }
    }

    const int rowBase = mBase + half * 8;   // lanes 0-15 -> M 0..7, lanes 16-31 -> M 8..15
#pragma unroll
    for (int nt = 0; nt < 8; nt++) {
        const int col = nBase + nt * 16 + l16;
        const float bv = bias ? bias[col] : 0.0f;
#pragma unroll
        for (int r = 0; r < 8; r++)
            C[(size_t)(rowBase + r) * N + col] = acc[nt][r] + bv;
    }
}

// ---------------- reductions ----------------
__device__ __forceinline__ float warpSum(float v) {
#pragma unroll
    for (int off = 16; off > 0; off >>= 1) v += __shfl_xor(v, off, 32);
    return v;
}
__device__ __forceinline__ float warpMax(float v) {
#pragma unroll
    for (int off = 16; off > 0; off >>= 1) v = fmaxf(v, __shfl_xor(v, off, 32));
    return v;
}

// ---------------- elementwise / conversion kernels ----------------
__global__ __launch_bounds__(256) void k_f32_to_bf16(__bf16* dst, const float* src, int n) {
    int i = blockIdx.x * 256 + threadIdx.x;
    if (i < n) dst[i] = (__bf16)src[i];
}
__global__ __launch_bounds__(256) void k_f32_to_bf16_padrows(__bf16* dst, const float* src,
                                                             int Nsrc, int Npad, int K) {
    int i = blockIdx.x * 256 + threadIdx.x;
    if (i >= Npad * K) return;
    int n = i / K, k = i - n * K;
    dst[i] = (n < Nsrc) ? (__bf16)src[(size_t)n * K + k] : (__bf16)0.0f;
}
__global__ __launch_bounds__(256) void k_add_vec(float* dst, const float* a, const float* b, int n) {
    int i = blockIdx.x * 256 + threadIdx.x;
    if (i < n) dst[i] = a[i] + b[i];
}
__global__ __launch_bounds__(256) void k_pad_vec(float* dst, const float* src, int n, int npad) {
    int i = blockIdx.x * 256 + threadIdx.x;
    if (i < npad) dst[i] = (i < n) ? src[i] : 0.0f;
}
__global__ __launch_bounds__(256) void k_zero_f32(float* dst, int n) {
    int i = blockIdx.x * 256 + threadIdx.x;
    if (i < n) dst[i] = 0.0f;
}
__global__ __launch_bounds__(256) void k_copy_f32(float* dst, const float* src, int n) {
    int i = blockIdx.x * 256 + threadIdx.x;
    if (i < n) dst[i] = src[i];
}

// x_cat[b, 0:512] = embed[tok[b]], x_cat[b, 512:1536] = feed[b]   (bf16)
__global__ __launch_bounds__(256) void k_build_x(__bf16* x, const float* embed,
                                                 const int* tok_t, const float* feed) {
    int i = blockIdx.x * 256 + threadIdx.x;
    if (i >= Bb * (Ee + Hh)) return;
    int b = i / (Ee + Hh), c = i - b * (Ee + Hh);
    float v = (c < Ee) ? embed[(size_t)tok_t[b] * Ee + c] : feed[(size_t)b * Hh + (c - Ee)];
    x[i] = (__bf16)v;
}

// all_feat = [hidden, sum_ext, sum_self]  (bf16, B x 3H)
__global__ __launch_bounds__(256) void k_concat3(__bf16* out, const float* a,
                                                 const float* b_, const float* c_) {
    int i = blockIdx.x * 256 + threadIdx.x;
    if (i >= Bb * 3 * Hh) return;
    int b = i / (3 * Hh), c = i - b * 3 * Hh;
    float v = (c < Hh) ? a[(size_t)b * Hh + c]
            : (c < 2 * Hh) ? b_[(size_t)b * Hh + c - Hh]
            : c_[(size_t)b * Hh + c - 2 * Hh];
    out[i] = (__bf16)v;
}

// gates (B x 4H, [i|f|g|o]) -> c_new, h_new (f32 + bf16)
__global__ __launch_bounds__(256) void k_lstm_pw(const float* __restrict__ gates,
                                                 float* __restrict__ c_state,
                                                 float* __restrict__ h_out,
                                                 __bf16* __restrict__ h_bf) {
    int i = blockIdx.x * 256 + threadIdx.x;
    if (i >= Bb * Hh) return;
    int b = i >> 10, hh = i & (Hh - 1);
    const float* g = gates + (size_t)b * 4 * Hh;
    float ig = g[hh], fg = g[Hh + hh], gg = g[2 * Hh + hh], og = g[3 * Hh + hh];
    float si = 1.0f / (1.0f + expf(-ig));
    float sf = 1.0f / (1.0f + expf(-fg));
    float so = 1.0f / (1.0f + expf(-og));
    float cn = sf * c_state[i] + si * tanhf(gg);
    float hn = so * tanhf(cn);
    c_state[i] = cn;
    h_out[i] = hn;
    h_bf[i] = (__bf16)hn;
}

// attention: one block per batch row b. scores over Slen, softmax, weighted sum.
// tcur < 0: external (mask = token==PAD). tcur >= 0: self (mask = s<=t ? pad : 1).
__global__ __launch_bounds__(256) void k_attend(float* __restrict__ sum_out,
                                                float* __restrict__ dist_out,
                                                const float* __restrict__ feats,
                                                const float* __restrict__ q,
                                                const int* __restrict__ tokens,
                                                int Slen, int tcur) {
    const int b = blockIdx.x, tid = threadIdx.x;
    __shared__ float sc[64];
    __shared__ float red[8];
    float qv[4];
#pragma unroll
    for (int j = 0; j < 4; j++) qv[j] = q[(size_t)b * Hh + tid + j * 256];

    for (int s = 0; s < Slen; s++) {
        const float* f = feats + ((size_t)s * Bb + b) * Hh;
        float p = 0.0f;
#pragma unroll
        for (int j = 0; j < 4; j++) p += f[tid + j * 256] * qv[j];
        p = warpSum(p);
        if ((tid & 31) == 0) red[tid >> 5] = p;
        __syncthreads();
        if (tid == 0) {
            float t = 0.0f;
            for (int w = 0; w < 8; w++) t += red[w];
            float mask;
            if (tcur < 0) mask = (tokens[s * Bb + b] == 0) ? 1.0f : 0.0f;
            else mask = (s <= tcur) ? ((tokens[s * Bb + b] == 0) ? 1.0f : 0.0f) : 1.0f;
            sc[s] = t - NEGC * mask;
        }
        __syncthreads();
    }
    if (tid == 0) {
        float m = -3.4e38f;
        for (int s = 0; s < Slen; s++) m = fmaxf(m, sc[s]);
        float sum = 0.0f;
        for (int s = 0; s < Slen; s++) { float e = expf(sc[s] - m); sc[s] = e; sum += e; }
        float inv = 1.0f / sum;
        for (int s = 0; s < Slen; s++) sc[s] *= inv;
    }
    __syncthreads();
    if (tid < Slen) dist_out[tid * Bb + b] = sc[tid];

    float acc[4] = {0.f, 0.f, 0.f, 0.f};
    for (int s = 0; s < Slen; s++) {
        const float* f = feats + ((size_t)s * Bb + b) * Hh;
        float d = sc[s];
#pragma unroll
        for (int j = 0; j < 4; j++) acc[j] += d * f[tid + j * 256];
    }
#pragma unroll
    for (int j = 0; j < 4; j++) sum_out[(size_t)b * Hh + tid + j * 256] = acc[j];
}

// seq_probs = softmax(hidden @ W_seq^T + b_seq) over 2 classes; one block per b
__global__ __launch_bounds__(256) void k_seq(float* seq_probs, const float* hid,
                                             const float* Wseq, const float* bseq) {
    const int b = blockIdx.x, tid = threadIdx.x;
    __shared__ float red[16];
    float p0 = 0.f, p1 = 0.f;
    for (int i = tid; i < Hh; i += 256) {
        float hv = hid[(size_t)b * Hh + i];
        p0 += hv * Wseq[i];
        p1 += hv * Wseq[Hh + i];
    }
    p0 = warpSum(p0); p1 = warpSum(p1);
    if ((tid & 31) == 0) { red[tid >> 5] = p0; red[8 + (tid >> 5)] = p1; }
    __syncthreads();
    if (tid == 0) {
        float a = bseq[0], c = bseq[1];
        for (int w = 0; w < 8; w++) { a += red[w]; c += red[8 + w]; }
        float m = fmaxf(a, c);
        float e0 = expf(a - m), e1 = expf(c - m);
        float inv = 1.0f / (e0 + e1);
        seq_probs[b * 2 + 0] = e0 * inv;
        seq_probs[b * 2 + 1] = e1 * inv;
    }
}

// pred softmax + copy mechanism + log + adjustments. One block per b.
__global__ __launch_bounds__(256) void k_final(float* __restrict__ out,
                                               const float* __restrict__ predlog,
                                               const float* __restrict__ seq_probs,
                                               const float* __restrict__ dist_ext,
                                               const float* __restrict__ dist_self,
                                               const int* __restrict__ att_tokens,
                                               const int* __restrict__ ref_tokens,
                                               int t, int is_last) {
    const int b = blockIdx.x, tid = threadIdx.x;
    __shared__ float probs[VPAD];
    __shared__ float copyp[VPAD];
    __shared__ float red[8];

    float lmax = -3.4e38f;
    for (int v = tid; v < VPAD; v += 256) {
        float x = (v < Vv) ? predlog[(size_t)b * VPAD + v] : -3.4e38f;
        probs[v] = x;
        lmax = fmaxf(lmax, (v < Vv) ? x : -3.4e38f);
    }
    lmax = warpMax(lmax);
    if ((tid & 31) == 0) red[tid >> 5] = lmax;
    __syncthreads();
    float m = red[0];
    for (int w = 1; w < 8; w++) m = fmaxf(m, red[w]);
    __syncthreads();
    float lsum = 0.0f;
    for (int v = tid; v < Vv; v += 256) { float e = expf(probs[v] - m); probs[v] = e; lsum += e; }
    lsum = warpSum(lsum);
    __syncthreads();
    if ((tid & 31) == 0) red[tid >> 5] = lsum;
    __syncthreads();
    float ssum = 0.0f;
    for (int w = 0; w < 8; w++) ssum += red[w];
    float inv = 1.0f / ssum;
    for (int v = tid; v < Vv; v += 256) { probs[v] *= inv; copyp[v] = EPSC; }
    __syncthreads();

    const float copy_w = probs[4];  // COPY = 4
    const float wext  = seq_probs[b * 2 + 0] * copy_w;
    const float wself = seq_probs[b * 2 + 1] * copy_w;
    for (int s = tid; s < Ss; s += 256)
        atomicAdd(&copyp[att_tokens[s * Bb + b]], dist_ext[s * Bb + b] * wext);
    for (int s = tid; s < Tt; s += 256)
        atomicAdd(&copyp[ref_tokens[s * Bb + b]], dist_self[s * Bb + b] * wself);
    __syncthreads();

    for (int v = tid; v < Vv; v += 256) {
        float comb = copyp[v] + probs[v];
        if (v == 4) comb = 1.0f;
        float lg = logf(comb);
        if (v == 4) lg = -1000000000.0f;
        if (v <= 3) lg -= NEGC;            // base_adj on [PAD..UNK]
        if (v == 2) lg += NEGC;            // base_adj EOS (+NEG)
        if (is_last && v == 2) lg += NEGC; // eos_adj at t == MAXLEN-1
        out[((size_t)t * Bb + b) * Vv + v] = lg;
    }
}

// ---------------- host ----------------
#define CDIV(a, b) (((a) + (b) - 1) / (b))

extern "C" void kernel_launch(void* const* d_in, const int* in_sizes, int n_in,
                              void* d_out, int out_size, void* d_ws, size_t ws_size,
                              hipStream_t stream) {
    const float* embed    = (const float*)d_in[0];
    const float* W_ih0    = (const float*)d_in[1];
    const float* W_hh0    = (const float*)d_in[2];
    const float* b_ih0    = (const float*)d_in[3];
    const float* b_hh0    = (const float*)d_in[4];
    const float* W_ih1    = (const float*)d_in[5];
    const float* W_hh1    = (const float*)d_in[6];
    const float* b_ih1    = (const float*)d_in[7];
    const float* b_hh1    = (const float*)d_in[8];
    const float* Wq_ext   = (const float*)d_in[9];
    const float* bq_ext   = (const float*)d_in[10];
    const float* Wq_self  = (const float*)d_in[11];
    const float* bq_self  = (const float*)d_in[12];
    const float* W_comb   = (const float*)d_in[13];
    const float* b_comb   = (const float*)d_in[14];
    const float* W_pred   = (const float*)d_in[15];
    const float* b_pred   = (const float*)d_in[16];
    const float* W_seq    = (const float*)d_in[17];
    const float* b_seq    = (const float*)d_in[18];
    const float* att_feat = (const float*)d_in[19];
    const float* h_init   = (const float*)d_in[20];
    const float* c_init   = (const float*)d_in[21];
    const int*   ref_tok  = (const int*)d_in[22];
    const int*   att_tok  = (const int*)d_in[23];
    float* out = (float*)d_out;

    // workspace carve-out
    char* base = (char*)d_ws;
    size_t off = 0;
    auto alloc = [&](size_t bytes) -> void* {
        off = (off + 255) & ~(size_t)255;
        void* p = base + off;
        off += bytes;
        return p;
    };
    const int BH = Bb * Hh;
    __bf16* wih0b  = (__bf16*)alloc((size_t)4 * Hh * (Ee + Hh) * 2);
    __bf16* whh0b  = (__bf16*)alloc((size_t)4 * Hh * Hh * 2);
    __bf16* wih1b  = (__bf16*)alloc((size_t)4 * Hh * Hh * 2);
    __bf16* whh1b  = (__bf16*)alloc((size_t)4 * Hh * Hh * 2);
    __bf16* wqeb   = (__bf16*)alloc((size_t)Hh * Hh * 2);
    __bf16* wqsb   = (__bf16*)alloc((size_t)Hh * Hh * 2);
    __bf16* wcombb = (__bf16*)alloc((size_t)Hh * 3 * Hh * 2);
    __bf16* wpredb = (__bf16*)alloc((size_t)VPAD * Hh * 2);
    float*  bias0  = (float*)alloc(4 * Hh * 4);
    float*  bias1  = (float*)alloc(4 * Hh * 4);
    float*  bpredp = (float*)alloc(VPAD * 4);
    float*  feed   = (float*)alloc((size_t)BH * 4);          // comb / feed
    __bf16* xcat   = (__bf16*)alloc((size_t)Bb * (Ee + Hh) * 2);
    float*  gates  = (float*)alloc((size_t)Bb * 4 * Hh * 4);
    float*  h0f    = (float*)alloc((size_t)BH * 4);
    float*  c0f    = (float*)alloc((size_t)BH * 4);
    float*  c1f    = (float*)alloc((size_t)BH * 4);
    __bf16* h0b    = (__bf16*)alloc((size_t)BH * 2);
    __bf16* h1b    = (__bf16*)alloc((size_t)BH * 2);
    float*  hidbuf = (float*)alloc((size_t)Tt * BH * 4);
    float*  qe     = (float*)alloc((size_t)BH * 4);
    float*  qs     = (float*)alloc((size_t)BH * 4);
    float*  se     = (float*)alloc((size_t)BH * 4);
    float*  ssm    = (float*)alloc((size_t)BH * 4);
    float*  de     = (float*)alloc((size_t)Ss * Bb * 4);
    float*  dsf    = (float*)alloc((size_t)Tt * Bb * 4);
    __bf16* allf   = (__bf16*)alloc((size_t)Bb * 3 * Hh * 2);
    __bf16* combb  = (__bf16*)alloc((size_t)BH * 2);
    float*  plog   = (float*)alloc((size_t)Bb * VPAD * 4);
    float*  sqp    = (float*)alloc((size_t)Bb * 2 * 4);

    // ---- prologue: weight conversion (bf16), bias fusion, state init ----
    auto conv = [&](__bf16* d, const float* s, int n) {
        k_f32_to_bf16<<<CDIV(n, 256), 256, 0, stream>>>(d, s, n);
    };
    conv(wih0b, W_ih0, 4 * Hh * (Ee + Hh));
    conv(whh0b, W_hh0, 4 * Hh * Hh);
    conv(wih1b, W_ih1, 4 * Hh * Hh);
    conv(whh1b, W_hh1, 4 * Hh * Hh);
    conv(wqeb,  Wq_ext, Hh * Hh);
    conv(wqsb,  Wq_self, Hh * Hh);
    conv(wcombb, W_comb, Hh * 3 * Hh);
    k_f32_to_bf16_padrows<<<CDIV(VPAD * Hh, 256), 256, 0, stream>>>(wpredb, W_pred, Vv, VPAD, Hh);
    k_add_vec<<<CDIV(4 * Hh, 256), 256, 0, stream>>>(bias0, b_ih0, b_hh0, 4 * Hh);
    k_add_vec<<<CDIV(4 * Hh, 256), 256, 0, stream>>>(bias1, b_ih1, b_hh1, 4 * Hh);
    k_pad_vec<<<CDIV(VPAD, 256), 256, 0, stream>>>(bpredp, b_pred, Vv, VPAD);
    k_zero_f32<<<CDIV(BH, 256), 256, 0, stream>>>(feed, BH);
    k_zero_f32<<<CDIV(Tt * BH, 256), 256, 0, stream>>>(hidbuf, Tt * BH);
    k_copy_f32<<<CDIV(BH, 256), 256, 0, stream>>>(c0f, c_init, BH);
    k_copy_f32<<<CDIV(BH, 256), 256, 0, stream>>>(c1f, c_init + BH, BH);
    conv(h0b, h_init, BH);
    conv(h1b, h_init + BH, BH);

    const dim3 blk(256);
    const dim3 g4096(4 * Hh / 128, Bb / 128);   // (32, 2)
    const dim3 g1024(Hh / 128, Bb / 128);       // (8, 2)

    for (int t = 0; t < Tt; t++) {
        float* hid_t = hidbuf + (size_t)t * BH;

        // x = [embed[ref[t]], feed]
        k_build_x<<<CDIV(Bb * (Ee + Hh), 256), blk, 0, stream>>>(xcat, embed, ref_tok + t * Bb, feed);

        // LSTM layer 0: gates = x @ W_ih0^T + h0 @ W_hh0^T + (b_ih0 + b_hh0)
        gemm2_bf16_wmma<<<g4096, blk, 0, stream>>>(gates, xcat, wih0b, Ee + Hh,
                                                   h0b, whh0b, Hh, bias0, 4 * Hh);
        k_lstm_pw<<<CDIV(BH, 256), blk, 0, stream>>>(gates, c0f, h0f, h0b);

        // LSTM layer 1 (hidden written straight into hidbuf[t])
        gemm2_bf16_wmma<<<g4096, blk, 0, stream>>>(gates, h0b, wih1b, Hh,
                                                   h1b, whh1b, Hh, bias1, 4 * Hh);
        k_lstm_pw<<<CDIV(BH, 256), blk, 0, stream>>>(gates, c1f, hid_t, h1b);

        // attention queries
        gemm2_bf16_wmma<<<g1024, blk, 0, stream>>>(qe, h1b, wqeb, Hh,
                                                   nullptr, nullptr, 0, bq_ext, Hh);
        gemm2_bf16_wmma<<<g1024, blk, 0, stream>>>(qs, h1b, wqsb, Hh,
                                                   nullptr, nullptr, 0, bq_self, Hh);

        // external + self attention
        k_attend<<<Bb, blk, 0, stream>>>(se, de, att_feat, qe, att_tok, Ss, -1);
        k_attend<<<Bb, blk, 0, stream>>>(ssm, dsf, hidbuf, qs, ref_tok, Tt, t);

        // comb = [hidden, sum_ext, sum_self] @ W_comb^T + b_comb  (becomes next feed)
        k_concat3<<<CDIV(Bb * 3 * Hh, 256), blk, 0, stream>>>(allf, hid_t, se, ssm);
        gemm2_bf16_wmma<<<g1024, blk, 0, stream>>>(feed, allf, wcombb, 3 * Hh,
                                                   nullptr, nullptr, 0, b_comb, Hh);

        // pred logits
        k_f32_to_bf16<<<CDIV(BH, 256), blk, 0, stream>>>(combb, feed, BH);
        gemm2_bf16_wmma<<<g1024, blk, 0, stream>>>(plog, combb, wpredb, Hh,
                                                   nullptr, nullptr, 0, bpredp, VPAD);

        // seq gate + final copy/softmax/log
        k_seq<<<Bb, blk, 0, stream>>>(sqp, hid_t, W_seq, b_seq);
        k_final<<<Bb, blk, 0, stream>>>(out, plog, sqp, de, dsf, att_tok, ref_tok,
                                        t, (t == Tt - 1) ? 1 : 0);
    }
    (void)in_sizes; (void)n_in; (void)out_size; (void)ws_size;
}
